// OldDeepSetsEnsemble_58145267253430
// MI455X (gfx1250) — compile-verified, matrix operand order checked by reading, not run on hardware
//
#include <hip/hip_runtime.h>
#include <hip/hip_bf16.h>
#include <cmath>

// ---------- types for WMMA fragments ----------
typedef __attribute__((ext_vector_type(16))) __bf16 v16bf;
typedef __attribute__((ext_vector_type(8)))  __bf16 v8bf;
typedef __attribute__((ext_vector_type(2)))  __bf16 v2bf;
typedef __attribute__((ext_vector_type(8)))  float  v8f;

union Frag16 { v16bf v; v8bf h[2]; };

#define T_TOTAL 524288
#define B_EV    4096
#define NPTS    256     // points per block (one segment pair)
#define HSTRIDE 136     // halfwords per point-row in LDS (128 + 8 pad; 272B = 17x16B aligned)
#define XSTRIDE 40      // halfwords per point-row of the padded K=32 input tile (80B = 5x16B)

__device__ __forceinline__ float relu_f(float v) { return v > 0.f ? v : 0.f; }

// ============================================================
// Kernel 0: weight prep (L2-resident afterwards):
//   w1,w2 (128x128 f32) -> bf16
//   pw0   (128x8  f32)  -> bf16 zero-padded to 128x32 (K=8..31 = 0)
// ============================================================
__global__ void cvt_weights(const float* __restrict__ w0f,
                            const float* __restrict__ w1f,
                            const float* __restrict__ w2f,
                            __bf16* __restrict__ w0p,
                            __bf16* __restrict__ w1b,
                            __bf16* __restrict__ w2b) {
    int i = blockIdx.x * blockDim.x + threadIdx.x;
    if (i < 128 * 128) {
        w1b[i] = (__bf16)w1f[i];
        w2b[i] = (__bf16)w2f[i];
    }
    if (i < 128 * 32) {
        const int m = i >> 5, k = i & 31;
        w0p[i] = (k < 8) ? (__bf16)w0f[m * 8 + k] : (__bf16)0.f;
    }
}

// ============================================================
// Templated bf16 WMMA GEMM layer: out[128 x 256] = relu(W @ in + b)
// Each wave owns 32 points (two 16-col N tiles), loops 8 M tiles.
// Fragments: per lane two 16B chunks, matching the CDNA5 16-bit
// A/B VGPR layout (lane group g selects the +8 K-half).
// ============================================================
template<int NKS>
__device__ __forceinline__ void gemm_layer(const __bf16* __restrict__ W, int kstride,
                                           const float* __restrict__ bias,
                                           const __bf16* hin, int instride,
                                           __bf16* hout,
                                           int colbase, int ln, int g) {
    for (int mt = 0; mt < 8; ++mt) {
        v8f acc0 = {}; v8f acc1 = {};
#pragma unroll
        for (int ks = 0; ks < NKS; ++ks) {
            const __bf16* wr = W + (size_t)(mt * 16 + ln) * kstride + ks * 32 + g * 8;
            Frag16 a; a.h[0] = *(const v8bf*)wr; a.h[1] = *(const v8bf*)(wr + 16);
            const __bf16* bp0 = hin + (colbase + ln) * instride + ks * 32 + g * 8;
            Frag16 b0; b0.h[0] = *(const v8bf*)bp0; b0.h[1] = *(const v8bf*)(bp0 + 16);
            const __bf16* bp1 = bp0 + 16 * instride;
            Frag16 b1; b1.h[0] = *(const v8bf*)bp1; b1.h[1] = *(const v8bf*)(bp1 + 16);

            acc0 = __builtin_amdgcn_wmma_f32_16x16x32_bf16(
                       false, a.v, false, b0.v, (short)0, acc0, false, false);
            acc1 = __builtin_amdgcn_wmma_f32_16x16x32_bf16(
                       false, a.v, false, b1.v, (short)0, acc1, false, false);
        }
        // epilogue: bias + ReLU + bf16 pack; C/D layout: lane ln = col N,
        // element r -> M = r + 8*g_half
        const int col0 = colbase + ln;
        const int col1 = col0 + 16;
#pragma unroll
        for (int r = 0; r < 8; r += 2) {
            const int m = mt * 16 + r + 8 * g;
            v2bf p0, p1;
            p0[0] = (__bf16)relu_f(acc0[r]     + bias[m]);
            p0[1] = (__bf16)relu_f(acc0[r + 1] + bias[m + 1]);
            p1[0] = (__bf16)relu_f(acc1[r]     + bias[m]);
            p1[1] = (__bf16)relu_f(acc1[r + 1] + bias[m + 1]);
            *(v2bf*)(hout + col0 * HSTRIDE + m) = p0;
            *(v2bf*)(hout + col1 * HSTRIDE + m) = p1;
        }
    }
}

// ============================================================
// Kernel 1: fused phi (all 3 layers bf16 WMMA) + segment mean
// pooling + concat. One block = one segment pair = 256 points.
// ============================================================
__global__ __launch_bounds__(256) void phi_pool(
    const float* __restrict__ x_set,                       // [8][T]
    const __bf16* __restrict__ w0p, const float* __restrict__ pb0,  // [128][32] padded
    const __bf16* __restrict__ w1,  const float* __restrict__ pb1,  // [128][128]
    const __bf16* __restrict__ w2,  const float* __restrict__ pb2,
    const float* __restrict__ x_flat,                      // [4096][16]
    const int*  __restrict__ counts,                       // sample_indices row 0
    float* __restrict__ z)                                 // [4096][144]
{
    __shared__ __bf16 hbuf[2][NPTS * HSTRIDE];   // ping-pong activation tiles
    __shared__ float2 partials[256];             // pooling combine scratch

    const int blk  = blockIdx.x;
    const int t    = threadIdx.x;
    const int base = blk * NPTS;

    // ---------- stage input tile: x[point][K=32 padded] bf16 into hbuf[1] ----------
    {
        v8bf xr = {};
#pragma unroll
        for (int c = 0; c < 8; ++c)
            xr[c] = (__bf16)x_set[(size_t)c * T_TOTAL + base + t];
        v8bf zero = {};
        __bf16* row = &hbuf[1][t * XSTRIDE];
        *(v8bf*)(row)      = xr;    // K 0..7  (real channels)
        *(v8bf*)(row + 8)  = zero;  // K 8..15
        *(v8bf*)(row + 16) = zero;  // K 16..23
        *(v8bf*)(row + 24) = zero;  // K 24..31
    }
    __syncthreads();

    const int wave = t >> 5;
    const int lane = t & 31;
    const int g    = lane >> 4;
    const int ln   = lane & 15;
    const int colbase = wave * 32;

    // layer 0: K=32 (padded), in = x tile (hbuf[1], XSTRIDE), out = hbuf[0]
    gemm_layer<1>(w0p, 32, pb0, hbuf[1], XSTRIDE, hbuf[0], colbase, ln, g);
    __syncthreads();
    // layer 1: in = hbuf[0], out = hbuf[1] (overwrites x tile; already consumed)
    gemm_layer<4>(w1, 128, pb1, hbuf[0], HSTRIDE, hbuf[1], colbase, ln, g);
    __syncthreads();
    // layer 2: in = hbuf[1], out = hbuf[0]
    gemm_layer<4>(w2, 128, pb2, hbuf[1], HSTRIDE, hbuf[0], colbase, ln, g);
    __syncthreads();

    // ---------- segment mean pooling ----------
    // two threads per (segment, channel-pair): each sums every other column
    // with a single b32 (v2bf) LDS load per 2 channels.
    const __bf16* h2 = hbuf[0];
    const int c0 = counts[2 * blk];
    const int c1 = counts[2 * blk + 1];
    const int combo   = t >> 1;          // 0..127
    const int halfidx = t & 1;
    const int seg = combo >> 6;          // 0..1
    const int ch  = (combo & 63) * 2;    // 0,2,...,126
    const int lo  = seg ? c0 : 0;
    int hi        = seg ? (c0 + c1) : c0;
    if (hi > NPTS) hi = NPTS;

    float s0 = 0.f, s1 = 0.f;
    for (int col = lo + halfidx; col < hi; col += 2) {
        v2bf p = *(const v2bf*)(h2 + col * HSTRIDE + ch);
        s0 += (float)p[0];
        s1 += (float)p[1];
    }
    partials[t] = make_float2(s0, s1);
    __syncthreads();
    if (halfidx == 0) {
        const float2 o = partials[t + 1];
        const float inv = 1.f / (float)(hi - lo);
        const size_t b = 2 * blk + seg;
        z[b * 144 + ch]     = (s0 + o.x) * inv;
        z[b * 144 + ch + 1] = (s1 + o.y) * inv;
    }

    // concat flat features
    if (t < 32) {
        const int bb = 2 * blk + (t >> 4);
        const int j  = t & 15;
        z[(size_t)bb * 144 + 128 + j] = x_flat[bb * 16 + j];
    }
}

// ============================================================
// Head kernels (tiny: ~0.6 GFLOP, ~12 MB traffic -> fp32 VALU)
// ============================================================
__global__ __launch_bounds__(256) void mlp0(const float* __restrict__ z,
                                            const float* __restrict__ mw0,
                                            const float* __restrict__ mb0,
                                            float* __restrict__ z1) {
    __shared__ float zr[144];
    const int b = blockIdx.x;
    for (int i = threadIdx.x; i < 144; i += 256) zr[i] = z[(size_t)b * 144 + i];
    __syncthreads();
    const int o = threadIdx.x;
    const float* w = mw0 + (size_t)o * 144;
    float acc = mb0[o];
    for (int k = 0; k < 144; ++k) acc = fmaf(w[k], zr[k], acc);
    z1[(size_t)b * 256 + o] = acc;
}

__global__ void bn_stats(const float* __restrict__ x, int F,
                         float* __restrict__ mu, float* __restrict__ rstd) {
    const int f = threadIdx.x;
    if (f >= F) return;
    float s = 0.f, s2 = 0.f;
    for (int b = 0; b < B_EV; ++b) {
        float v = x[(size_t)b * F + f];
        s += v; s2 += v * v;
    }
    const float m   = s  * (1.f / B_EV);
    const float var = s2 * (1.f / B_EV) - m * m;
    mu[f]   = m;
    rstd[f] = rsqrtf(var + 1e-5f);
}

__global__ __launch_bounds__(128) void mlp1(const float* __restrict__ z1,
                                            const float* __restrict__ mw1,
                                            const float* __restrict__ mb1,
                                            const float* __restrict__ g0,
                                            const float* __restrict__ be0,
                                            const float* __restrict__ mu0,
                                            const float* __restrict__ rs0,
                                            float* __restrict__ z2) {
    __shared__ float a[256];
    const int b = blockIdx.x;
    for (int i = threadIdx.x; i < 256; i += 128) {
        float v = z1[(size_t)b * 256 + i];
        v = g0[i] * (v - mu0[i]) * rs0[i] + be0[i];
        a[i] = relu_f(v);
    }
    __syncthreads();
    const int o = threadIdx.x;
    const float* w = mw1 + (size_t)o * 256;
    float acc = mb1[o];
    for (int k = 0; k < 256; ++k) acc = fmaf(w[k], a[k], acc);
    z2[(size_t)b * 128 + o] = acc;
}

__global__ __launch_bounds__(256) void head_out(const float* __restrict__ z2,
                                                const float* __restrict__ mw2,
                                                const float* __restrict__ mb2,
                                                const float* __restrict__ g1,
                                                const float* __restrict__ be1,
                                                const float* __restrict__ mu1,
                                                const float* __restrict__ rs1,
                                                float* __restrict__ out) {
    const int b = blockIdx.x * blockDim.x + threadIdx.x;
    if (b >= B_EV) return;
    float acc = mb2[0];
    for (int k = 0; k < 128; ++k) {
        float v = z2[(size_t)b * 128 + k];
        v = g1[k] * (v - mu1[k]) * rs1[k] + be1[k];
        acc = fmaf(mw2[k], relu_f(v), acc);
    }
    out[b] = 1.f / (1.f + __expf(-acc));
}

// ============================================================
// launch
// ============================================================
extern "C" void kernel_launch(void* const* d_in, const int* in_sizes, int n_in,
                              void* d_out, int out_size, void* d_ws, size_t ws_size,
                              hipStream_t stream) {
    const float* x_set  = (const float*)d_in[0];
    const float* x_flat = (const float*)d_in[1];
    const float* pw0    = (const float*)d_in[2];
    const float* pb0    = (const float*)d_in[3];
    const float* pw1    = (const float*)d_in[4];
    const float* pb1    = (const float*)d_in[5];
    const float* pw2    = (const float*)d_in[6];
    const float* pb2    = (const float*)d_in[7];
    const float* mw0    = (const float*)d_in[8];
    const float* mb0    = (const float*)d_in[9];
    const float* g0     = (const float*)d_in[10];
    const float* be0    = (const float*)d_in[11];
    const float* mw1    = (const float*)d_in[12];
    const float* mb1    = (const float*)d_in[13];
    const float* g1     = (const float*)d_in[14];
    const float* be1    = (const float*)d_in[15];
    const float* mw2    = (const float*)d_in[16];
    const float* mb2    = (const float*)d_in[17];
    const int*   sidx   = (const int*)  d_in[18];   // row 0 = counts[4096]

    // workspace layout
    char* ws = (char*)d_ws;
    size_t o = 0;
    __bf16* w0p = (__bf16*)(ws + o); o += 128 * 32  * sizeof(__bf16);          // 8 KB
    __bf16* w1b = (__bf16*)(ws + o); o += 128 * 128 * sizeof(__bf16);          // 32 KB
    __bf16* w2b = (__bf16*)(ws + o); o += 128 * 128 * sizeof(__bf16);          // 32 KB
    float*  z   = (float*)(ws + o);  o += (size_t)B_EV * 144 * sizeof(float);  // 2.25 MB
    float*  z1  = (float*)(ws + o);  o += (size_t)B_EV * 256 * sizeof(float);  // 4 MB
    float*  z2  = (float*)(ws + o);  o += (size_t)B_EV * 128 * sizeof(float);  // 2 MB
    float*  mu0 = (float*)(ws + o);  o += 256 * sizeof(float);
    float*  rs0 = (float*)(ws + o);  o += 256 * sizeof(float);
    float*  mu1 = (float*)(ws + o);  o += 128 * sizeof(float);
    float*  rs1 = (float*)(ws + o);  o += 128 * sizeof(float);

    float* out = (float*)d_out;

    cvt_weights<<<(128 * 128 + 255) / 256, 256, 0, stream>>>(pw0, pw1, pw2,
                                                             w0p, w1b, w2b);

    phi_pool<<<B_EV / 2, 256, 0, stream>>>(x_set, w0p, pb0, w1b, pb1, w2b, pb2,
                                           x_flat, sidx, z);

    mlp0<<<B_EV, 256, 0, stream>>>(z, mw0, mb0, z1);
    bn_stats<<<1, 256, 0, stream>>>(z1, 256, mu0, rs0);
    mlp1<<<B_EV, 128, 0, stream>>>(z1, mw1, mb1, g0, be0, mu0, rs0, z2);
    bn_stats<<<1, 128, 0, stream>>>(z2, 128, mu1, rs1);
    head_out<<<(B_EV + 255) / 256, 256, 0, stream>>>(z2, mw2, mb2, g1, be1, mu1, rs1, out);
}
